// MambaTransformerHybrid_22952305230032
// MI455X (gfx1250) — matrix operand order repaired and checked by reading, not compiled, not run on hardware
//
#include <hip/hip_runtime.h>

// ---------------------------------------------------------------------------
// Types / helpers for CDNA5 WMMA (wave32, 16x16x32 bf16 -> f32)
// ---------------------------------------------------------------------------
typedef __attribute__((ext_vector_type(16))) __bf16 v16bf;
typedef __attribute__((ext_vector_type(8)))  __bf16 v8bf;
typedef __attribute__((ext_vector_type(8)))  float  v8f;

#define CAT16(lo, hi) __builtin_shufflevector((lo), (hi), 0,1,2,3,4,5,6,7,8,9,10,11,12,13,14,15)
// (neg_a, A, neg_b, B, c_mod, C, reuse_a, reuse_b)
#define WMMA_BF16(a, b, c) \
  __builtin_amdgcn_wmma_f32_16x16x32_bf16(false, (a), false, (b), (short)0, (c), false, false)

__device__ __forceinline__ __bf16 f2bf(float f) {
  unsigned u = __builtin_bit_cast(unsigned, f);
  unsigned r = (u + 0x7FFFu + ((u >> 16) & 1u)) >> 16;   // round-to-nearest-even
  return __builtin_bit_cast(__bf16, (unsigned short)r);
}
__device__ __forceinline__ float sigmoidf_(float v) { return 1.f / (1.f + __expf(-v)); }

// ---------------------------------------------------------------------------
// Model dims (compile-time)
// ---------------------------------------------------------------------------
#define BSZ     2
#define SEQ     1024
#define DM      1024      // D_MODEL
#define DI      2048      // D_INNER
#define DS      16        // D_STATE
#define DFF     4096
#define NROWS   (BSZ * SEQ)     // 2048
#define EPSF    1.1920929e-07f

// ---------------------------------------------------------------------------
// WMMA GEMM: C[M,N] = epilogue(A[M,K] @ W[K,N])
// block tile 64x128, wave tile 32x32 (2x2 frags of 16x16), K-step 32
// EPI: 0=none 1=+bias 2=softplus(+bias) 3=silu 4=+R 5=+bias+R 6=*R
// ---------------------------------------------------------------------------
template <int EPI>
__global__ __launch_bounds__(256) void gemm_wmma(
    const float* __restrict__ A, const float* __restrict__ W,
    float* __restrict__ C, const float* __restrict__ bias,
    const float* __restrict__ R, int M, int N, int K) {
  __shared__ __bf16 sA[64][32];     // row-major A tile
  __shared__ __bf16 sBt[128][32];   // B tile stored transposed: sBt[n][k]

  const int tid   = threadIdx.x;
  const int lane  = tid & 31;
  const int wave  = tid >> 5;
  const int mbase = blockIdx.y * 64;
  const int nbase = blockIdx.x * 128;
  const int wm    = (wave & 1) * 32;
  const int wn    = (wave >> 1) * 32;

  const v8f vz = {0.f, 0.f, 0.f, 0.f, 0.f, 0.f, 0.f, 0.f};
  v8f acc[2][2];
  acc[0][0] = vz; acc[0][1] = vz; acc[1][0] = vz; acc[1][1] = vz;

  const int arow = tid >> 2, ak0 = (tid & 3) * 8;   // 64 rows x 32 k, 8 f32/thread
  const int bkk  = tid >> 3, bn0 = (tid & 7) * 16;  // 32 k   x 128 n, 16 f32/thread

  for (int k0 = 0; k0 < K; k0 += 32) {
    __syncthreads();
    {
      const float* ap = A + (size_t)(mbase + arow) * K + k0 + ak0;
#pragma unroll
      for (int j = 0; j < 8; ++j) sA[arow][ak0 + j] = f2bf(ap[j]);
      const float* wp = W + (size_t)(k0 + bkk) * N + nbase + bn0;
#pragma unroll
      for (int j = 0; j < 16; ++j) sBt[bn0 + j][bkk] = f2bf(wp[j]);
      if (k0 + 32 < K) {                      // gfx1250 global_prefetch_b8
        __builtin_prefetch(ap + 32, 0, 1);
        __builtin_prefetch(W + (size_t)(k0 + 32 + bkk) * N + nbase + bn0, 0, 1);
      }
    }
    __syncthreads();

    // A frag: lane<16 -> K 0..7 & 16..23 ; lane>=16 -> K 8..15 & 24..31
    v16bf af[2], bfr[2];
#pragma unroll
    for (int mi = 0; mi < 2; ++mi) {
      int row = wm + mi * 16 + (lane & 15);
      int kb  = (lane >> 4) * 8;
      v8bf lo = *(const v8bf*)&sA[row][kb];
      v8bf hi = *(const v8bf*)&sA[row][kb + 16];
      af[mi] = CAT16(lo, hi);
    }
    // B frag: lane<16 -> K 0..7 & 16..23 ; lane>=16 -> K 8..15 & 24..31
#pragma unroll
    for (int ni = 0; ni < 2; ++ni) {
      int col = wn + ni * 16 + (lane & 15);
      int kb  = (lane >> 4) * 8;
      v8bf lo = *(const v8bf*)&sBt[col][kb];
      v8bf hi = *(const v8bf*)&sBt[col][kb + 16];
      bfr[ni] = CAT16(lo, hi);
    }
#pragma unroll
    for (int mi = 0; mi < 2; ++mi)
#pragma unroll
      for (int ni = 0; ni < 2; ++ni)
        acc[mi][ni] = WMMA_BF16(af[mi], bfr[ni], acc[mi][ni]);
  }

  // C layout: lane<16 rows m0..m0+7, lane>=16 rows m0+8..m0+15; col = lane&15
#pragma unroll
  for (int mi = 0; mi < 2; ++mi) {
#pragma unroll
    for (int ni = 0; ni < 2; ++ni) {
      int row0 = mbase + wm + mi * 16 + (lane >> 4) * 8;
      int col  = nbase + wn + ni * 16 + (lane & 15);
#pragma unroll
      for (int r = 0; r < 8; ++r) {
        size_t idx = (size_t)(row0 + r) * N + col;
        float v = acc[mi][ni][r];
        if (EPI == 1 || EPI == 2 || EPI == 5) v += bias[col];
        if (EPI == 2) v = (v > 20.f) ? v : __logf(1.f + __expf(v));   // softplus
        if (EPI == 3) v = v * sigmoidf_(v);                           // silu
        if (EPI == 4 || EPI == 5) v += R[idx];
        if (EPI == 6) v *= R[idx];
        C[idx] = v;
      }
    }
  }
}

// ---------------------------------------------------------------------------
// Embedding gather: x[b,t,:] = emb[ids[b,t],:]
// ---------------------------------------------------------------------------
__global__ __launch_bounds__(256) void k_embed(const int* __restrict__ ids,
                                               const float* __restrict__ emb,
                                               float* __restrict__ x) {
  int i   = blockIdx.x * 256 + threadIdx.x;   // NROWS*DM
  int col = i & (DM - 1);
  int row = i >> 10;
  x[i] = emb[(size_t)ids[row] * DM + col];
}

// ---------------------------------------------------------------------------
// RMSNorm per row (D elements), safe in-place
// ---------------------------------------------------------------------------
__global__ __launch_bounds__(256) void k_rmsnorm(const float* __restrict__ in,
                                                 const float* __restrict__ w,
                                                 float* __restrict__ out, int D) {
  __shared__ float red[256];
  int row = blockIdx.x;
  const float* r = in + (size_t)row * D;
  float s = 0.f;
  for (int c = threadIdx.x; c < D; c += 256) { float v = r[c]; s += v * v; }
  red[threadIdx.x] = s;
  __syncthreads();
  for (int st = 128; st > 0; st >>= 1) {
    if (threadIdx.x < st) red[threadIdx.x] += red[threadIdx.x + st];
    __syncthreads();
  }
  float scale = rsqrtf(red[0] / (float)D + EPSF);
  for (int c = threadIdx.x; c < D; c += 256)
    out[(size_t)row * D + c] = r[c] * scale * w[c];
}

// ---------------------------------------------------------------------------
// Depthwise causal conv(width 4) + bias + SiLU.  Input = xres[:, :DI] (stride 2*DI)
// ---------------------------------------------------------------------------
__global__ __launch_bounds__(256) void k_conv_silu(const float* __restrict__ xres,
                                                   const float* __restrict__ cw,
                                                   const float* __restrict__ cb,
                                                   float* __restrict__ xc) {
  int i = blockIdx.x * 256 + threadIdx.x;    // BSZ*SEQ*DI
  int d = i & (DI - 1);
  int t = (i >> 11) & (SEQ - 1);
  int b = i >> 21;
  float acc = cb[d];
#pragma unroll
  for (int j = 0; j < 4; ++j) {
    int tt = t - 3 + j;
    if (tt >= 0) acc += cw[d * 4 + j] * xres[(size_t)(b * SEQ + tt) * (2 * DI) + d];
  }
  xc[i] = acc * sigmoidf_(acc);
}

// ---------------------------------------------------------------------------
// Bc = xc @ xp[:, DS:2*DS]   (K=2048, N=16)  — one block per row
// ---------------------------------------------------------------------------
__global__ __launch_bounds__(256) void k_bproj(const float* __restrict__ xc,
                                               const float* __restrict__ xp,
                                               float* __restrict__ Bc) {
  __shared__ float sm[256][DS];
  int row = blockIdx.x, tid = threadIdx.x;
  float part[DS];
#pragma unroll
  for (int s = 0; s < DS; ++s) part[s] = 0.f;
  for (int d = tid; d < DI; d += 256) {
    float v = xc[(size_t)row * DI + d];
    const float* wr = xp + (size_t)d * (2 * DS) + DS;
#pragma unroll
    for (int s = 0; s < DS; ++s) part[s] += v * wr[s];
  }
#pragma unroll
  for (int s = 0; s < DS; ++s) sm[tid][s] = part[s];
  __syncthreads();
  if (tid < DS) {
    float acc = 0.f;
    for (int t = 0; t < 256; ++t) acc += sm[t][tid];
    Bc[(size_t)row * DS + tid] = acc;
  }
}

// ---------------------------------------------------------------------------
// Selective scan, fused with  y = (scan + xc*D) * silu(res).  1 thread/(b,d)
// ---------------------------------------------------------------------------
__global__ __launch_bounds__(256) void k_scan(const float* __restrict__ xc,
                                              const float* __restrict__ delta,
                                              const float* __restrict__ Bc,
                                              const float* __restrict__ Alog,
                                              const float* __restrict__ Dp,
                                              const float* __restrict__ xres,
                                              float* __restrict__ y) {
  int c = blockIdx.x * 256 + threadIdx.x;    // BSZ*DI
  int b = c >> 11;
  int d = c & (DI - 1);
  float Ac[DS], h[DS];
#pragma unroll
  for (int s = 0; s < DS; ++s) { Ac[s] = -__expf(Alog[d * DS + s]); h[s] = 0.f; }
  float Dd = Dp[d];
  for (int t = 0; t < SEQ; ++t) {
    size_t row = (size_t)(b * SEQ + t);
    float xt = xc[row * DI + d];
    float dt = delta[row * DI + d];
    const float* bc = Bc + row * DS;
    float sum = 0.f;
#pragma unroll
    for (int s = 0; s < DS; ++s) {
      h[s] = __expf(dt * Ac[s]) * h[s] + dt * bc[s] * xt;
      sum += h[s];
    }
    float rv = xres[row * (2 * DI) + DI + d];
    y[row * DI + d] = (sum + xt * Dd) * (rv * sigmoidf_(rv));
  }
}

// ---------------------------------------------------------------------------
// Flash attention (WMMA): block = (b, head, 128-query tile), 8 waves,
// each wave owns 16 queries; stream 64-key tiles with online softmax.
// qkv layout: [b, t, {q,k,v}, head, 64]
// ---------------------------------------------------------------------------
__global__ __launch_bounds__(256) void k_attn(const float* __restrict__ qkv,
                                              float* __restrict__ out) {
  constexpr int QT = 128, KT = 64, HD = 64;
  __shared__ __bf16 sQ[QT][HD];        // 16 KB
  __shared__ __bf16 sK[KT][HD];        //  8 KB (key-major)
  __shared__ __bf16 sVt[HD][KT];       //  8 KB (transposed: [d][key])
  __shared__ __bf16 sP[8][16][KT];     // 16 KB (per-wave P tile)

  const int tid = threadIdx.x, lane = tid & 31, wave = tid >> 5;
  const int qbase = blockIdx.x * QT;
  const int h = blockIdx.y, b = blockIdx.z;

  // load Q tile (which=0)
  for (int i = tid; i < QT * HD; i += 256) {
    int qq = i >> 6, d = i & 63;
    sQ[qq][d] = f2bf(qkv[((size_t)(b * SEQ + qbase + qq) * 3) * DM + h * HD + d]);
  }
  __syncthreads();

  // per-wave Q fragments (A-matrix, 2 K-steps over head dim)
  v16bf aQ[2];
  {
    int row = wave * 16 + (lane & 15);
#pragma unroll
    for (int ks = 0; ks < 2; ++ks) {
      int kb = ks * 32 + (lane >> 4) * 8;
      v8bf lo = *(const v8bf*)&sQ[row][kb];
      v8bf hi = *(const v8bf*)&sQ[row][kb + 16];
      aQ[ks] = CAT16(lo, hi);
    }
  }

  const v8f vz = {0.f, 0.f, 0.f, 0.f, 0.f, 0.f, 0.f, 0.f};
  float m_i[8], l_i[8];
  v8f o[4];
#pragma unroll
  for (int r = 0; r < 8; ++r) { m_i[r] = -3.0e38f; l_i[r] = 0.f; }
#pragma unroll
  for (int di = 0; di < 4; ++di) o[di] = vz;

  for (int kt = 0; kt < SEQ / KT; ++kt) {
    int kb2 = kt * KT;
    __syncthreads();
    for (int i = tid; i < KT * HD; i += 256) {
      int kk = i >> 6, d = i & 63;
      size_t base = (size_t)(b * SEQ + kb2 + kk) * 3 * DM + h * HD + d;
      sK[kk][d]  = f2bf(qkv[base + DM]);       // which=1 (K)
      sVt[d][kk] = f2bf(qkv[base + 2 * DM]);   // which=2 (V, transposed)
    }
    __syncthreads();

    // S strip (16 queries x 64 keys) = 4 n-frags, K-dim = head dim (2 steps)
    v8f s[4];
#pragma unroll
    for (int ni = 0; ni < 4; ++ni) s[ni] = vz;
#pragma unroll
    for (int ks = 0; ks < 2; ++ks) {
#pragma unroll
      for (int ni = 0; ni < 4; ++ni) {
        int key = ni * 16 + (lane & 15);
        int kb  = ks * 32 + (lane >> 4) * 8;
        v8bf lo = *(const v8bf*)&sK[key][kb];
        v8bf hi = *(const v8bf*)&sK[key][kb + 16];
        v16bf bK = CAT16(lo, hi);
        s[ni] = WMMA_BF16(aQ[ks], bK, s[ni]);
      }
    }
#pragma unroll
    for (int ni = 0; ni < 4; ++ni) s[ni] = s[ni] * 0.125f;   // 1/sqrt(64)

    // online softmax; row = wave*16 + (lane>>4)*8 + r, spread over 16 lanes
#pragma unroll
    for (int r = 0; r < 8; ++r) {
      float rmax = fmaxf(fmaxf(s[0][r], s[1][r]), fmaxf(s[2][r], s[3][r]));
      for (int off = 1; off < 16; off <<= 1) rmax = fmaxf(rmax, __shfl_xor(rmax, off, 32));
      float mnew  = fmaxf(m_i[r], rmax);
      float alpha = __expf(m_i[r] - mnew);
      m_i[r] = mnew;
      float p[4], rsum = 0.f;
#pragma unroll
      for (int ni = 0; ni < 4; ++ni) { p[ni] = __expf(s[ni][r] - mnew); rsum += p[ni]; }
      for (int off = 1; off < 16; off <<= 1) rsum += __shfl_xor(rsum, off, 32);
      l_i[r] = l_i[r] * alpha + rsum;
#pragma unroll
      for (int di = 0; di < 4; ++di) o[di][r] *= alpha;
      int prow = (lane >> 4) * 8 + r;
#pragma unroll
      for (int ni = 0; ni < 4; ++ni) sP[wave][prow][ni * 16 + (lane & 15)] = f2bf(p[ni]);
    }

    // O += P @ V  (K-dim = keys, 2 steps of 32)
#pragma unroll
    for (int ks = 0; ks < 2; ++ks) {
      int arow = lane & 15;
      int akb  = ks * 32 + (lane >> 4) * 8;
      v8bf alo = *(const v8bf*)&sP[wave][arow][akb];
      v8bf ahi = *(const v8bf*)&sP[wave][arow][akb + 16];
      v16bf aP = CAT16(alo, ahi);
#pragma unroll
      for (int di = 0; di < 4; ++di) {
        int col = di * 16 + (lane & 15);
        int kb  = ks * 32 + (lane >> 4) * 8;
        v8bf blo = *(const v8bf*)&sVt[col][kb];
        v8bf bhi = *(const v8bf*)&sVt[col][kb + 16];
        v16bf bV = CAT16(blo, bhi);
        o[di] = WMMA_BF16(aP, bV, o[di]);
      }
    }
  }

  // write O / l
#pragma unroll
  for (int r = 0; r < 8; ++r) {
    int q = qbase + wave * 16 + (lane >> 4) * 8 + r;
    float inv = 1.f / l_i[r];
#pragma unroll
    for (int di = 0; di < 4; ++di)
      out[(size_t)(b * SEQ + q) * DM + h * HD + di * 16 + (lane & 15)] = o[di][r] * inv;
  }
}

// ---------------------------------------------------------------------------
// Host launcher
// ---------------------------------------------------------------------------
extern "C" void kernel_launch(void* const* d_in, const int* in_sizes, int n_in,
                              void* d_out, int out_size, void* d_ws, size_t ws_size,
                              hipStream_t stream) {
  (void)in_sizes; (void)n_in; (void)out_size; (void)ws_size;
  const int*   ids      = (const int*)d_in[0];
  const float* emb      = (const float*)d_in[1];
  const float* m_norm   = (const float*)d_in[2];
  const float* m_inp    = (const float*)d_in[3];
  const float* m_convw  = (const float*)d_in[4];
  const float* m_convb  = (const float*)d_in[5];
  const float* m_xp     = (const float*)d_in[6];
  const float* m_dtw    = (const float*)d_in[7];
  const float* m_dtb    = (const float*)d_in[8];
  const float* m_Alog   = (const float*)d_in[9];
  const float* m_D      = (const float*)d_in[10];
  const float* m_outp   = (const float*)d_in[11];
  const float* t_n1     = (const float*)d_in[12];
  const float* t_n2     = (const float*)d_in[13];
  const float* t_qkv    = (const float*)d_in[14];
  const float* t_ow     = (const float*)d_in[15];
  const float* t_ob     = (const float*)d_in[16];
  const float* t_w1     = (const float*)d_in[17];
  const float* t_w2     = (const float*)d_in[18];
  const float* t_w3     = (const float*)d_in[19];
  const float* fin_norm = (const float*)d_in[20];
  const float* lm_head  = (const float*)d_in[21];
  float* logits = (float*)d_out;

  // workspace arena (floats); total ~31.5M floats (~126 MB)
  float* ws    = (float*)d_ws;
  float* x     = ws;                       // 2048*1024
  float* xr    = ws + 2097152;             // 2048*1024 (also attention out)
  float* xres  = ws + 4194304;             // 2048*4096 (also FFN h1)
  float* xc    = ws + 12582912;            // 2048*2048
  float* delta = ws + 16777216;            // 2048*2048
  float* ybuf  = ws + 20971520;            // 2048*2048
  float* Bc    = ws + 25165824;            // 2048*16
  float* qkvb  = ws + 25198592;            // 2048*3072

  dim3 blk(256);
  k_embed<<<NROWS * DM / 256, blk, 0, stream>>>(ids, emb, x);

  for (int i = 0; i < 4; ++i) {
    // ---- Mamba block ----
    k_rmsnorm<<<NROWS, blk, 0, stream>>>(x, m_norm + i * DM, xr, DM);
    gemm_wmma<0><<<dim3(2 * DI / 128, NROWS / 64), blk, 0, stream>>>(
        xr, m_inp + (size_t)i * DM * 2 * DI, xres, nullptr, nullptr, NROWS, 2 * DI, DM);
    k_conv_silu<<<NROWS * DI / 256, blk, 0, stream>>>(
        xres, m_convw + (size_t)i * DI * 4, m_convb + i * DI, xc);
    k_bproj<<<NROWS, blk, 0, stream>>>(xc, m_xp + (size_t)i * DI * 2 * DS, Bc);
    gemm_wmma<2><<<dim3(DI / 128, NROWS / 64), blk, 0, stream>>>(
        xc, m_dtw + (size_t)i * DI * DI, delta, m_dtb + i * DI, nullptr, NROWS, DI, DI);
    k_scan<<<BSZ * DI / 256, blk, 0, stream>>>(
        xc, delta, Bc, m_Alog + (size_t)i * DI * DS, m_D + i * DI, xres, ybuf);
    gemm_wmma<4><<<dim3(DM / 128, NROWS / 64), blk, 0, stream>>>(
        ybuf, m_outp + (size_t)i * DI * DM, x, nullptr, x, NROWS, DM, DI);

    // ---- Transformer block ----
    gemm_wmma<0><<<dim3(3 * DM / 128, NROWS / 64), blk, 0, stream>>>(
        x, t_qkv + (size_t)i * DM * 3 * DM, qkvb, nullptr, nullptr, NROWS, 3 * DM, DM);
    k_attn<<<dim3(SEQ / 128, 16, BSZ), blk, 0, stream>>>(qkvb, xr);
    gemm_wmma<5><<<dim3(DM / 128, NROWS / 64), blk, 0, stream>>>(
        xr, t_ow + (size_t)i * DM * DM, x, t_ob + i * DM, x, NROWS, DM, DM);
    k_rmsnorm<<<NROWS, blk, 0, stream>>>(x, t_n1 + i * DM, x, DM);
    gemm_wmma<3><<<dim3(DFF / 128, NROWS / 64), blk, 0, stream>>>(
        x, t_w1 + (size_t)i * DM * DFF, xres, nullptr, nullptr, NROWS, DFF, DM);
    gemm_wmma<6><<<dim3(DFF / 128, NROWS / 64), blk, 0, stream>>>(
        x, t_w2 + (size_t)i * DM * DFF, xres, nullptr, xres, NROWS, DFF, DM);
    gemm_wmma<4><<<dim3(DM / 128, NROWS / 64), blk, 0, stream>>>(
        xres, t_w3 + (size_t)i * DFF * DM, x, nullptr, x, NROWS, DM, DFF);
    k_rmsnorm<<<NROWS, blk, 0, stream>>>(x, t_n2 + i * DM, x, DM);
  }

  k_rmsnorm<<<NROWS, blk, 0, stream>>>(x, fin_norm, xr, DM);
  gemm_wmma<0><<<dim3(32000 / 128, NROWS / 64), blk, 0, stream>>>(
      xr, lm_head, logits, nullptr, nullptr, NROWS, 32000, DM);
}
